// VectorQuantizer_31568009626252
// MI455X (gfx1250) — compile-verified, matrix operand order checked by reading, not run on hardware
//
#include <hip/hip_runtime.h>

#define B_ROWS  32768
#define LATENT_ 1024
#define EMB     32
#define NUMZ    8192
#define TILE_B  128
#define NWAVE   8
#define NBLK    (B_ROWS / TILE_B)   // 256
#define ZC      256                 // codes per LDS chunk (double-buffered)
#define NCH     (NUMZ / ZC)         // 32 chunks

typedef __attribute__((ext_vector_type(2)))  float  v2f;
typedef __attribute__((ext_vector_type(8)))  float  v8f;
typedef __attribute__((ext_vector_type(8)))  __bf16 v8bf;
typedef __attribute__((ext_vector_type(16))) __bf16 v16bf;

// ---- workspace layout (bytes) ----
#define WS_WN_OFF  0u                          // bf16 wn [8192*32]  = 512 KB
#define WS_NW_OFF  (NUMZ * EMB * 2u)           // f32 ||wn||^2 [8192] = 32 KB
#define WS_ACC_OFF (WS_NW_OFF + NUMZ * 4u)     // f32 per-block loss partials [256]

// ---- d_out layout (float elements), reference return order ----
#define LOSS_OFF ((size_t)B_ROWS * LATENT_)    // 33554432
#define IDX_OFF  (LOSS_OFF + 3)
#define ZE_OFF   (IDX_OFF + B_ROWS)

// ---------------------------------------------------------------------------
// Prep: wn = l2norm(codebook) in bf16, nw = sum(wn*wn). One wave per code row.
// ---------------------------------------------------------------------------
__global__ __launch_bounds__(256) void vq_prep(const float* __restrict__ cb,
                                               __bf16* __restrict__ wn,
                                               float* __restrict__ nw) {
    int row  = (blockIdx.x * blockDim.x + threadIdx.x) >> 5;
    int lane = threadIdx.x & 31;
    if (row >= NUMZ) return;
    float v  = cb[(size_t)row * EMB + lane];
    float ss = v * v;
#pragma unroll
    for (int m = 16; m >= 1; m >>= 1) ss += __shfl_xor(ss, m, 32);
    float scale = 1.f / fmaxf(sqrtf(ss), 1e-12f);
    wn[(size_t)row * EMB + lane] = (__bf16)(v * scale);
    if (lane == 0) nw[row] = ss * scale * scale;
}

// ---------------------------------------------------------------------------
// Fused main kernel: in_proj (f32 WMMA) -> l2norm -> NN search (bf16 WMMA,
// async double-buffered codebook staging) -> gather/loss -> out_proj (f32 WMMA)
// ---------------------------------------------------------------------------
__global__ __launch_bounds__(256) void vq_main(const float* __restrict__ z_e,
                                               const float* __restrict__ in_w,
                                               const float* __restrict__ in_b,
                                               const float* __restrict__ out_w,
                                               const float* __restrict__ out_b,
                                               const float* __restrict__ cb,
                                               const __bf16* __restrict__ wn,
                                               const float* __restrict__ nw,
                                               float* __restrict__ partial,
                                               float* __restrict__ out) {
    __shared__ __align__(16) __bf16 s_wn[2][ZC * EMB];        // 2 x 16 KB
    __shared__ __align__(16) float  s_nw[2][ZC];              // 2 x  1 KB
    // overlay: bf16 zn (8 KB, dead after A-operand build) under f32 zq (16 KB)
    __shared__ __align__(16) char   s_mix[NWAVE * 16 * EMB * 4];
    __shared__ float s_ws[NWAVE];

    __bf16* s_zn = (__bf16*)s_mix;   // [NWAVE][16*EMB]
    float*  s_zq = (float*)s_mix;    // [NWAVE][16*EMB]

    const int tid  = threadIdx.x;
    const int w    = tid >> 5;
    const int lane = tid & 31;
    const int hf   = lane >> 4;      // which K/M half this lane serves
    const int l15  = lane & 15;
    const int R0   = blockIdx.x * TILE_B + w * 16;   // this wave's 16 rows

    // ---------------- stage 1: ze = z_e @ in_w.T + in_b (f32 WMMA) ----------
    v8f c0 = {}, c1 = {};            // N=0..15 and N=16..31 accumulators
    const float* arow = z_e  + (size_t)(R0 + l15) * LATENT_ + hf * 2;
    const float* b0p  = in_w + (size_t)l15        * LATENT_ + hf * 2;
    const float* b1p  = in_w + (size_t)(16 + l15) * LATENT_ + hf * 2;
    for (int k = 0; k < LATENT_; k += 4) {
        v2f a  = *(const v2f*)(arow + k);
        v2f b0 = *(const v2f*)(b0p + k);
        v2f b1 = *(const v2f*)(b1p + k);
        c0 = __builtin_amdgcn_wmma_f32_16x16x4_f32(false, a, false, b0, (short)0, c0, false, false);
        c1 = __builtin_amdgcn_wmma_f32_16x16x4_f32(false, a, false, b1, (short)0, c1, false, false);
    }
    const float bia0 = in_b[l15], bia1 = in_b[16 + l15];
#pragma unroll
    for (int r = 0; r < 8; r++) { c0[r] += bia0; c1[r] += bia1; }

    // row-wise 1/max(||ze||, eps) via 16-lane xor reduction (stays in half)
    float rn[8];
#pragma unroll
    for (int r = 0; r < 8; r++) {
        float t = c0[r] * c0[r] + c1[r] * c1[r];
        t += __shfl_xor(t, 1, 32); t += __shfl_xor(t, 2, 32);
        t += __shfl_xor(t, 4, 32); t += __shfl_xor(t, 8, 32);
        rn[r] = 1.f / fmaxf(sqrtf(t), 1e-12f);
    }
    // emit ze; stash bf16 zn row-major in LDS
#pragma unroll
    for (int r = 0; r < 8; r++) {
        int mrow = r + hf * 8;
        size_t gr = (size_t)(R0 + mrow);
        out[ZE_OFF + gr * EMB + l15]      = c0[r];
        out[ZE_OFF + gr * EMB + 16 + l15] = c1[r];
        s_zn[w * 16 * EMB + mrow * EMB + l15]      = (__bf16)(c0[r] * rn[r]);
        s_zn[w * 16 * EMB + mrow * EMB + 16 + l15] = (__bf16)(c1[r] * rn[r]);
    }
    __syncthreads();

    // build A operand (16x32 bf16): lane l<16 row M=l holds K 0..7,16..23;
    // lanes 16..31 hold K 8..15,24..31 -> two 16 B contiguous chunks per lane.
    v16bf a2;
    {
        const __bf16* zp = s_zn + w * 16 * EMB + l15 * EMB;
        v8bf lo = *(const v8bf*)(zp + hf * 8);
        v8bf hi = *(const v8bf*)(zp + 16 + hf * 8);
#pragma unroll
        for (int i = 0; i < 8; i++) { a2[i] = lo[i]; a2[8 + i] = hi[i]; }
    }

    // LDS byte offsets for the async staging (generic shared ptr low 32 bits)
    const unsigned lds_wn = (unsigned)(uintptr_t)&s_wn[0][0] + (unsigned)tid * 64u;
    const unsigned lds_nw = (unsigned)(uintptr_t)&s_nw[0][0] + (unsigned)tid * 4u;

    // issue one chunk of async global->LDS copies (5 ops/thread: 4xb128 + b32)
    auto issue_chunk = [&](int ci, int b) {
        unsigned lw = lds_wn + (unsigned)b * (ZC * EMB * 2u);
        unsigned gw = (unsigned)ci * (ZC * EMB * 2u) + (unsigned)tid * 64u;
        asm volatile("global_load_async_to_lds_b128 %0, %1, %2 offset:0"
                     :: "v"(lw), "v"(gw), "s"(wn) : "memory");
        asm volatile("global_load_async_to_lds_b128 %0, %1, %2 offset:16"
                     :: "v"(lw), "v"(gw), "s"(wn) : "memory");
        asm volatile("global_load_async_to_lds_b128 %0, %1, %2 offset:32"
                     :: "v"(lw), "v"(gw), "s"(wn) : "memory");
        asm volatile("global_load_async_to_lds_b128 %0, %1, %2 offset:48"
                     :: "v"(lw), "v"(gw), "s"(wn) : "memory");
        unsigned ln = lds_nw + (unsigned)b * (ZC * 4u);
        unsigned gn = (unsigned)ci * (ZC * 4u) + (unsigned)tid * 4u;
        asm volatile("global_load_async_to_lds_b32 %0, %1, %2 offset:0"
                     :: "v"(ln), "v"(gn), "s"(nw) : "memory");
    };

    // ---------------- stage 2: argmin over 8192 codes (bf16 WMMA) ----------
    float best[8]; int bi[8];
#pragma unroll
    for (int r = 0; r < 8; r++) { best[r] = 3.4e38f; bi[r] = 0; }

    issue_chunk(0, 0);
    for (int ci = 0; ci < NCH; ++ci) {
        asm volatile("s_wait_asynccnt 0" ::: "memory");  // my chunk-ci writes done
        __syncthreads();   // chunk ci visible; all waves done with chunk ci-1
        if (ci + 1 < NCH) issue_chunk(ci + 1, (ci + 1) & 1);  // overlap w/ compute

        const __bf16* wbuf = s_wn[ci & 1];
        const float*  nbuf = s_nw[ci & 1];
#pragma unroll 4
        for (int nt = 0; nt < ZC / 16; nt++) {
            const __bf16* wp = wbuf + (nt * 16 + l15) * EMB;
            v8bf lo = *(const v8bf*)(wp + hf * 8);
            v8bf hi = *(const v8bf*)(wp + 16 + hf * 8);
            v16bf b2;
#pragma unroll
            for (int i = 0; i < 8; i++) { b2[i] = lo[i]; b2[8 + i] = hi[i]; }
            v8f c = {};
            c = __builtin_amdgcn_wmma_f32_16x16x32_bf16(false, a2, false, b2,
                                                        (short)0, c, false, false);
            float nwv = nbuf[nt * 16 + l15];
            int   j   = ci * ZC + nt * 16 + l15;
#pragma unroll
            for (int r = 0; r < 8; r++) {
                // dist = ||zn||^2 - 2 s + ||wn||^2; row term constant -> drop
                float d = fmaf(-2.f, c[r], nwv);
                if (d < best[r]) { best[r] = d; bi[r] = j; }
            }
        }
    }
    // 16-lane argmin reduction (first-occurrence tie-break on smaller idx)
#pragma unroll
    for (int r = 0; r < 8; r++) {
#pragma unroll
        for (int m = 1; m < 16; m <<= 1) {
            float od = __shfl_xor(best[r], m, 32);
            int   oi = __shfl_xor(bi[r],  m, 32);
            if (od < best[r] || (od == best[r] && oi < bi[r])) { best[r] = od; bi[r] = oi; }
        }
    }
    if (l15 == 0) {
#pragma unroll
        for (int r = 0; r < 8; r++)
            out[IDX_OFF + R0 + r + hf * 8] = (float)bi[r];
    }

    // ------------- gather z_q, accumulate deterministic loss ---------------
    float lsum = 0.f;
#pragma unroll
    for (int r = 0; r < 8; r++) {
        int mrow = r + hf * 8;
        int j    = bi[r];
        float q0 = cb[(size_t)j * EMB + l15];
        float q1 = cb[(size_t)j * EMB + 16 + l15];
        s_zq[w * 16 * EMB + mrow * EMB + l15]      = q0;   // zn overlay is dead
        s_zq[w * 16 * EMB + mrow * EMB + 16 + l15] = q1;
        float e0 = c0[r] - q0, e1 = c1[r] - q1;
        lsum = fmaf(e0, e0, lsum);
        lsum = fmaf(e1, e1, lsum);
    }
#pragma unroll
    for (int m = 16; m >= 1; m >>= 1) lsum += __shfl_xor(lsum, m, 32);
    if (lane == 0) s_ws[w] = lsum;
    __syncthreads();
    if (tid == 0) {
        float s = 0.f;
#pragma unroll
        for (int i = 0; i < NWAVE; i++) s += s_ws[i];
        partial[blockIdx.x] = s;
    }

    // ---------------- stage 3: out = z_q @ out_w.T + out_b (f32 WMMA) ------
    v2f aK[8];
#pragma unroll
    for (int kk = 0; kk < 8; kk++)
        aK[kk] = *(const v2f*)&s_zq[w * 16 * EMB + l15 * EMB + kk * 4 + hf * 2];

    for (int n0 = 0; n0 < LATENT_; n0 += 16) {
        const float* bwp = out_w + (size_t)(n0 + l15) * EMB + hf * 2;
        v8f d = {};
#pragma unroll
        for (int kk = 0; kk < 8; kk++) {
            v2f b = *(const v2f*)(bwp + kk * 4);
            d = __builtin_amdgcn_wmma_f32_16x16x4_f32(false, aK[kk], false, b,
                                                      (short)0, d, false, false);
        }
        float bias = out_b[n0 + l15];
#pragma unroll
        for (int r = 0; r < 8; r++) {
            int mrow = r + hf * 8;
            out[(size_t)(R0 + mrow) * LATENT_ + n0 + l15] = d[r] + bias;
        }
    }
}

// ---------------------------------------------------------------------------
// Final: serial (deterministic) sum of 256 partials -> losses
// ---------------------------------------------------------------------------
__global__ void vq_finish(const float* __restrict__ partial, float* __restrict__ out) {
    if (threadIdx.x == 0 && blockIdx.x == 0) {
        float s = 0.f;
        for (int i = 0; i < NBLK; i++) s += partial[i];
        float mse = s / (float)((size_t)B_ROWS * EMB);
        out[LOSS_OFF + 0] = 1.0f * mse + 0.25f * mse;  // CB_COST*cb + COMMIT*enc
        out[LOSS_OFF + 1] = mse;                       // enc_loss
        out[LOSS_OFF + 2] = mse;                       // cb_loss
    }
}

extern "C" void kernel_launch(void* const* d_in, const int* in_sizes, int n_in,
                              void* d_out, int out_size, void* d_ws, size_t ws_size,
                              hipStream_t stream) {
    (void)in_sizes; (void)n_in; (void)out_size; (void)ws_size;
    const float* z_e   = (const float*)d_in[0];
    const float* in_w  = (const float*)d_in[1];
    const float* in_b  = (const float*)d_in[2];
    const float* out_w = (const float*)d_in[3];
    const float* out_b = (const float*)d_in[4];
    const float* cb    = (const float*)d_in[5];

    char*   ws = (char*)d_ws;
    __bf16* wn = (__bf16*)(ws + WS_WN_OFF);
    float*  nw = (float*)(ws + WS_NW_OFF);
    float*  pa = (float*)(ws + WS_ACC_OFF);
    float*  out = (float*)d_out;

    vq_prep<<<NUMZ / 8, 256, 0, stream>>>(cb, wn, nw);
    vq_main<<<NBLK, 256, 0, stream>>>(z_e, in_w, in_b, out_w, out_b, cb, wn, nw, pa, out);
    vq_finish<<<1, 32, 0, stream>>>(pa, out);
}
